// PluckerKernelAttention_60739427500732
// MI455X (gfx1250) — matrix operand order repaired and checked by reading, not compiled, not run on hardware
//
#include <hip/hip_runtime.h>
#include <hip/hip_bf16.h>

// ---------------------------------------------------------------------------
// PluckerKernelAttention on gfx1250 (MI455X), CDNA5 WMMA (bf16 in / f32 acc)
// B=2, T=2048, D=1024, H=16, d_head=64
// GEMMs: 2x2 register-blocked WMMA tiles. Flash attention: async-LDS staged V.
// ---------------------------------------------------------------------------

typedef __bf16 bf16_t;
typedef __attribute__((ext_vector_type(16))) __bf16 v16bf;
typedef __attribute__((ext_vector_type(8)))  float  v8f;
typedef __attribute__((__vector_size__(16))) int    v4i_t;

#define B_DIM 2
#define T_DIM 2048
#define D_DIM 1024
#define H_DIM 16
#define DH_DIM 64
#define BT_DIM (B_DIM * T_DIM)          // 4096
#define PL_SCALE 0.40824829046386301637f // 6^(-1/2)

#if __has_builtin(__builtin_amdgcn_global_load_async_to_lds_b128)
#define HAVE_ASYNC_LDS 1
#else
#define HAVE_ASYNC_LDS 0
#endif

// ----------------------- WMMA fragment loaders -----------------------------
// A-matrix 16x32 bf16 (ISA 7.12.2): lanes 0-15 & 16-31 both hold M=0..15;
// lane<16: v0..3 -> K 0..7, v4..7 -> K 16..23 ; lane>=16: +8.
__device__ __forceinline__ v16bf load_a_frag(const float* __restrict__ A, int ld,
                                             int mBase, int kBase, int kValid) {
  const int l = threadIdx.x & 31;
  const float* row = A + (long)(mBase + (l & 15)) * ld + kBase;
  const int khalf = (l >= 16) ? 8 : 0;
  v16bf a;
#pragma unroll
  for (int e = 0; e < 16; ++e) {
    const int v = e >> 1, p = e & 1;
    const int k = ((v < 4) ? (2 * v) : (8 + 2 * v)) + khalf + p;
    const float f = (k < kValid) ? row[k] : 0.0f;
    a[e] = (bf16_t)f;
  }
  return a;
}

// Same A layout, sourced from a 16x16 f32 tile in LDS (P matrix).
__device__ __forceinline__ v16bf load_a_frag_lds16(const float* sP) {
  const int l = threadIdx.x & 31;
  const float* row = sP + (l & 15) * 16;
  const int khalf = (l >= 16) ? 8 : 0;
  v16bf a;
#pragma unroll
  for (int e = 0; e < 16; ++e) {
    const int v = e >> 1, p = e & 1;
    const int k = ((v < 4) ? (2 * v) : (8 + 2 * v)) + khalf + p;
    const float f = (k < 16) ? row[k] : 0.0f;
    a[e] = (bf16_t)f;
  }
  return a;
}

// B-matrix 32x16 bf16: N = lane%16; same K order as A (lane<16: K 0..7,16..23; lane>=16: +8).
// Source row-major W[N,K] (weights): Bfrag(k,n) = W[nBase+n, kBase+k].
__device__ __forceinline__ v16bf load_b_frag_nk(const float* __restrict__ W, int ld,
                                                int nBase, int kBase, int kValid) {
  const int l = threadIdx.x & 31;
  const float* row = W + (long)(nBase + (l & 15)) * ld + kBase;
  const int khalf = (l >= 16) ? 8 : 0;
  v16bf b;
#pragma unroll
  for (int e = 0; e < 16; ++e) {
    const int k = khalf + ((e < 8) ? e : (e + 8));
    const float f = (k < kValid) ? row[k] : 0.0f;
    b[e] = (bf16_t)f;
  }
  return b;
}

// B-matrix fragment from a 16x64 f32 tile in LDS (the staged V tile):
// Bfrag(k,n) = sV[k*64 + nOff + n], K padded 16 -> 32 with zeros.
__device__ __forceinline__ v16bf load_b_frag_ldsV(const float* sV, int nOff) {
  const int l = threadIdx.x & 31;
  const int n = nOff + (l & 15);
  const int khalf = (l >= 16) ? 8 : 0;
  v16bf b;
#pragma unroll
  for (int e = 0; e < 16; ++e) {
    const int k = khalf + ((e < 8) ? e : (e + 8));
    const float f = (k < 16) ? sV[k * 64 + n] : 0.0f;
    b[e] = (bf16_t)f;
  }
  return b;
}

// ----------------------- generic GEMM: C = A * W^T + bias ------------------
// A[M,K] f32 row-major, W[N,K] f32 row-major, C[M,N] f32.
// Block = 128 threads = 4 waves; each wave computes a 32x32 C tile
// (2x2 WMMA register blocking -> 4 wmma per K-step for 4 fragment loads).
// Block tile = 64x64. Requires M%64==0, N%64==0, K%32==0 (true here).
__global__ __launch_bounds__(128)
void plucker_gemm_wmma(const float* __restrict__ A, const float* __restrict__ W,
                       const float* __restrict__ bias, float* __restrict__ C,
                       int M, int N, int K) {
  const int w  = threadIdx.x >> 5;
  const int wm = w & 1, wn = w >> 1;
  const int mBase = blockIdx.x * 64 + wm * 32;
  const int nBase = blockIdx.y * 64 + wn * 32;

  v8f acc00 = {}, acc01 = {}, acc10 = {}, acc11 = {};
  for (int kc = 0; kc < K; kc += 32) {
    __builtin_prefetch(A + (long)mBase * K + kc + 256, 0, 1);
    __builtin_prefetch(W + (long)nBase * K + kc + 256, 0, 1);
    const v16bf a0 = load_a_frag(A, K, mBase,      kc, 32);
    const v16bf a1 = load_a_frag(A, K, mBase + 16, kc, 32);
    const v16bf b0 = load_b_frag_nk(W, K, nBase,      kc, 32);
    const v16bf b1 = load_b_frag_nk(W, K, nBase + 16, kc, 32);
    acc00 = __builtin_amdgcn_wmma_f32_16x16x32_bf16(false, a0, false, b0, (short)0, acc00, false, false);
    acc01 = __builtin_amdgcn_wmma_f32_16x16x32_bf16(false, a0, false, b1, (short)0, acc01, false, false);
    acc10 = __builtin_amdgcn_wmma_f32_16x16x32_bf16(false, a1, false, b0, (short)0, acc10, false, false);
    acc11 = __builtin_amdgcn_wmma_f32_16x16x32_bf16(false, a1, false, b1, (short)0, acc11, false, false);
  }

  const int l  = threadIdx.x & 31;
  const int nl = l & 15;
  const int mh = (l >= 16) ? 8 : 0;
  const float bv0 = bias ? bias[nBase + nl]      : 0.0f;
  const float bv1 = bias ? bias[nBase + 16 + nl] : 0.0f;
#pragma unroll
  for (int v = 0; v < 8; ++v) {
    const long m0 = mBase + mh + v;
    const long m1 = m0 + 16;
    C[m0 * N + nBase + nl]      = acc00[v] + bv0;
    C[m0 * N + nBase + 16 + nl] = acc01[v] + bv1;
    C[m1 * N + nBase + nl]      = acc10[v] + bv0;
    C[m1 * N + nBase + 16 + nl] = acc11[v] + bv1;
  }
}

// ---------------- exterior products -> L_q, J*L_k (elementwise) ------------
__global__ __launch_bounds__(256)
void plucker_exterior(const float* __restrict__ Q1, const float* __restrict__ Q2,
                      const float* __restrict__ K1, const float* __restrict__ K2,
                      float* __restrict__ Lq, float* __restrict__ JLk) {
  const int idx = blockIdx.x * blockDim.x + threadIdx.x; // over BT*H
  if (idx >= BT_DIM * H_DIM) return;
  const int bt = idx / H_DIM, h = idx % H_DIM;
  const long base4 = (long)bt * (H_DIM * 4) + h * 4;

  float L[6];
  { // ---- q side ----
    const float a0 = Q1[base4 + 0], a1 = Q1[base4 + 1], a2 = Q1[base4 + 2], a3 = Q1[base4 + 3];
    const float b0 = Q2[base4 + 0], b1 = Q2[base4 + 1], b2 = Q2[base4 + 2], b3 = Q2[base4 + 3];
    L[0] = a0 * b1 - a1 * b0; L[1] = a0 * b2 - a2 * b0; L[2] = a0 * b3 - a3 * b0;
    L[3] = a1 * b2 - a2 * b1; L[4] = a1 * b3 - a3 * b1; L[5] = a2 * b3 - a3 * b2;
    float s = 0.f;
#pragma unroll
    for (int c = 0; c < 6; ++c) s += L[c] * L[c];
    const float inv = 1.0f / fmaxf(sqrtf(s), 1e-12f);
#pragma unroll
    for (int c = 0; c < 6; ++c) Lq[(long)idx * 6 + c] = L[c] * inv;
  }
  { // ---- k side (+ J permute: (L5,-L4,L3,L2,-L1,L0)) ----
    const float a0 = K1[base4 + 0], a1 = K1[base4 + 1], a2 = K1[base4 + 2], a3 = K1[base4 + 3];
    const float b0 = K2[base4 + 0], b1 = K2[base4 + 1], b2 = K2[base4 + 2], b3 = K2[base4 + 3];
    L[0] = a0 * b1 - a1 * b0; L[1] = a0 * b2 - a2 * b0; L[2] = a0 * b3 - a3 * b0;
    L[3] = a1 * b2 - a2 * b1; L[4] = a1 * b3 - a3 * b1; L[5] = a2 * b3 - a3 * b2;
    float s = 0.f;
#pragma unroll
    for (int c = 0; c < 6; ++c) s += L[c] * L[c];
    const float inv = 1.0f / fmaxf(sqrtf(s), 1e-12f);
    float* o = JLk + (long)idx * 6;
    o[0] =  L[5] * inv; o[1] = -L[4] * inv; o[2] =  L[3] * inv;
    o[3] =  L[2] * inv; o[4] = -L[1] * inv; o[5] =  L[0] * inv;
  }
}

// ---------------- causal flash attention, one wave per 16 q-rows -----------
// grid = (B*H, T/16), block = 32 threads (1 wave).
// V tile (16x64 f32 = 4 KB) is staged into LDS with coalesced 128-bit
// async-to-LDS transfers (ASYNCcnt) overlapped with the softmax math.
__global__ __launch_bounds__(32)
void plucker_flash_attn(const float* __restrict__ Lq, const float* __restrict__ JLk,
                        const float* __restrict__ Vbuf, float* __restrict__ Obuf) {
  const int bh = blockIdx.x;
  const int b = bh / H_DIM, h = bh % H_DIM;
  const int qb = blockIdx.y;
  const int l = threadIdx.x & 31;
  const int mh = (l >= 16) ? 8 : 0;

  const float* LqBase = Lq  + (long)b * T_DIM * H_DIM * 6 + h * 6;  // row stride H*6
  const float* JkBase = JLk + (long)b * T_DIM * H_DIM * 6 + h * 6;
  const float* Vbase  = Vbuf + (long)b * T_DIM * D_DIM + h * DH_DIM; // row stride D

  __shared__ float sP[16 * 16];
  __shared__ float sV[16 * 64];

  // A fragment of L_q for this q-tile (K = 6 channels, zero-padded to 32)
  const v16bf a_q = load_a_frag(LqBase, H_DIM * 6, qb * 16, 0, 6);

  v8f o0 = {}, o1 = {}, o2 = {}, o3 = {};
  float mrow[8], lrow[8];
#pragma unroll
  for (int v = 0; v < 8; ++v) { mrow[v] = -1e30f; lrow[v] = 0.0f; }

  for (int kb = 0; kb <= qb; ++kb) {
    // ---- stage V tile (16 rows x 64 cols) into LDS ----
#if HAVE_ASYNC_LDS
    asm volatile("s_wait_dscnt 0" ::: "memory");   // WAR vs prev iter's LDS reads
#pragma unroll
    for (int i = 0; i < 8; ++i) {
      const int idx = i * 32 + l;                  // float4 index 0..255
      const int krow = idx >> 4, c4 = idx & 15;    // 16 float4 per 64-col row
      const float* g = Vbase + (long)(kb * 16 + krow) * D_DIM + c4 * 4;
      __builtin_amdgcn_global_load_async_to_lds_b128(
          (__attribute__((address_space(1))) v4i_t*)g,
          (__attribute__((address_space(3))) v4i_t*)(sV + idx * 4), 0, 0);
    }
#else
    __syncthreads();
#pragma unroll
    for (int i = 0; i < 8; ++i) {
      const int idx = i * 32 + l;
      const int krow = idx >> 4, c4 = idx & 15;
      const float4 d = *(const float4*)(Vbase + (long)(kb * 16 + krow) * D_DIM + c4 * 4);
      *(float4*)(sV + idx * 4) = d;
    }
    __syncthreads();
#endif

    // ---- scores tile: S = Lq(16x6) * JLk^T(6x16) ----
    const v16bf b_k = load_b_frag_nk(JkBase, H_DIM * 6, kb * 16, 0, 6);
    v8f s = {};
    s = __builtin_amdgcn_wmma_f32_16x16x32_bf16(false, a_q, false, b_k,
                                                (short)0, s, false, false);

    const int n_global = kb * 16 + (l & 15);
    float p[8], alpha[8];
#pragma unroll
    for (int v = 0; v < 8; ++v) {
      const int m_global = qb * 16 + mh + v;
      float sv = s[v] * PL_SCALE;
      if (n_global > m_global) sv = -1e30f;   // causal mask
      p[v] = sv;
    }
    // ---- online softmax: row reductions across the 16-lane half ----
#pragma unroll
    for (int v = 0; v < 8; ++v) {
      float rm = p[v];
#pragma unroll
      for (int off = 1; off < 16; off <<= 1)
        rm = fmaxf(rm, __shfl_xor(rm, off, 16));
      const float mnew = fmaxf(mrow[v], rm);
      const float e = __expf(p[v] - mnew);
      p[v] = e;
      float rs = e;
#pragma unroll
      for (int off = 1; off < 16; off <<= 1)
        rs += __shfl_xor(rs, off, 16);
      alpha[v] = __expf(mrow[v] - mnew);
      lrow[v] = lrow[v] * alpha[v] + rs;
      mrow[v] = mnew;
    }

    // ---- transpose P (C-layout -> A-layout) through LDS ----
    __syncthreads();
#pragma unroll
    for (int v = 0; v < 8; ++v) sP[(mh + v) * 16 + (l & 15)] = p[v];
    __syncthreads();
    const v16bf a_p = load_a_frag_lds16(sP);

    // ---- rescale accumulator ----
#pragma unroll
    for (int v = 0; v < 8; ++v) {
      o0[v] *= alpha[v]; o1[v] *= alpha[v]; o2[v] *= alpha[v]; o3[v] *= alpha[v];
    }

#if HAVE_ASYNC_LDS
    asm volatile("s_wait_asynccnt 0" ::: "memory"); // V tile resident in LDS
#endif
    // ---- O += P * V (4 WMMAs over the 64-wide head) ----
    {
      v16bf bv;
      bv = load_b_frag_ldsV(sV,  0);
      o0 = __builtin_amdgcn_wmma_f32_16x16x32_bf16(false, a_p, false, bv, (short)0, o0, false, false);
      bv = load_b_frag_ldsV(sV, 16);
      o1 = __builtin_amdgcn_wmma_f32_16x16x32_bf16(false, a_p, false, bv, (short)0, o1, false, false);
      bv = load_b_frag_ldsV(sV, 32);
      o2 = __builtin_amdgcn_wmma_f32_16x16x32_bf16(false, a_p, false, bv, (short)0, o2, false, false);
      bv = load_b_frag_ldsV(sV, 48);
      o3 = __builtin_amdgcn_wmma_f32_16x16x32_bf16(false, a_p, false, bv, (short)0, o3, false, false);
    }
    __syncthreads();   // protect sP before next iteration's writes
  }

  // ---- normalize and write O tile (16 x 64) ----
  float* Oout = Obuf + (long)b * T_DIM * D_DIM + h * DH_DIM;
#pragma unroll
  for (int v = 0; v < 8; ++v) {
    const float inv = 1.0f / lrow[v];
    const long r = (long)(qb * 16 + mh + v) * D_DIM + (l & 15);
    Oout[r +  0] = o0[v] * inv;
    Oout[r + 16] = o1[v] * inv;
    Oout[r + 32] = o2[v] * inv;
    Oout[r + 48] = o3[v] * inv;
  }
}

// ---------------------------------------------------------------------------
extern "C" void kernel_launch(void* const* d_in, const int* in_sizes, int n_in,
                              void* d_out, int out_size, void* d_ws, size_t ws_size,
                              hipStream_t stream) {
  const float* x    = (const float*)d_in[0];
  const float* W1q  = (const float*)d_in[1];
  const float* W2q  = (const float*)d_in[2];
  const float* W1k  = (const float*)d_in[3];
  const float* W2k  = (const float*)d_in[4];
  const float* Wv   = (const float*)d_in[5];
  const float* bv   = (const float*)d_in[6];
  const float* Wout = (const float*)d_in[7];
  const float* bout = (const float*)d_in[8];
  float* out = (float*)d_out;

  char* ws = (char*)d_ws;
  float* Q1   = (float*)(ws + (0ull  << 20));  // 4096 x 64
  float* Q2   = (float*)(ws + (1ull  << 20));
  float* K1   = (float*)(ws + (2ull  << 20));
  float* K2   = (float*)(ws + (3ull  << 20));
  float* Lq   = (float*)(ws + (4ull  << 20));  // 4096 x 16 x 6
  float* JLk  = (float*)(ws + (6ull  << 20));
  float* Vbuf = (float*)(ws + (8ull  << 20));  // 4096 x 1024
  float* Obuf = (float*)(ws + (24ull << 20));  // 4096 x 1024

  const dim3 blkG(128);
  // 1) projections: [4096,1024] x [64,1024]^T
  {
    const dim3 grd(BT_DIM / 64, 64 / 64);
    plucker_gemm_wmma<<<grd, blkG, 0, stream>>>(x, W1q, nullptr, Q1, BT_DIM, 64, D_DIM);
    plucker_gemm_wmma<<<grd, blkG, 0, stream>>>(x, W2q, nullptr, Q2, BT_DIM, 64, D_DIM);
    plucker_gemm_wmma<<<grd, blkG, 0, stream>>>(x, W1k, nullptr, K1, BT_DIM, 64, D_DIM);
    plucker_gemm_wmma<<<grd, blkG, 0, stream>>>(x, W2k, nullptr, K2, BT_DIM, 64, D_DIM);
  }
  // 2) exterior products -> normalized L_q and J*L_k
  {
    const int n = BT_DIM * H_DIM;
    plucker_exterior<<<(n + 255) / 256, 256, 0, stream>>>(Q1, Q2, K1, K2, Lq, JLk);
  }
  // 3) V = x @ W_v^T + b_v
  {
    const dim3 grd(BT_DIM / 64, D_DIM / 64);
    plucker_gemm_wmma<<<grd, blkG, 0, stream>>>(x, Wv, bv, Vbuf, BT_DIM, D_DIM, D_DIM);
  }
  // 4) causal flash attention
  {
    const dim3 grd(B_DIM * H_DIM, T_DIM / 16);
    plucker_flash_attn<<<grd, dim3(32), 0, stream>>>(Lq, JLk, Vbuf, Obuf);
  }
  // 5) out = O @ W_out^T + b_out
  {
    const dim3 grd(BT_DIM / 64, D_DIM / 64);
    plucker_gemm_wmma<<<grd, blkG, 0, stream>>>(Obuf, Wout, bout, out, BT_DIM, D_DIM, D_DIM);
  }
}
